// LstmAutoencoder_68358699483573
// MI455X (gfx1250) — matrix operand order repaired and checked by reading, not compiled, not run on hardware
//
#include <hip/hip_runtime.h>
#include <stdint.h>
#include <stddef.h>

// ---------------------------------------------------------------------------
// LSTM autoencoder for MI455X (gfx1250, wave32, WMMA).
//   encoder: 128 steps of  pre = x_t*w_ih^T + h*w_hh^T + b ; gates ; c,h
//   decoder: 128 steps of  pre = h*w_hh^T + b              ; gates ; c,h
//   output : softmax over a size-1 axis == 1.0 everywhere.
//
// GEMM path: v_wmma_f32_16x16x32_bf16, f32 accumulation.
// Cooperative block: 8 waves share one hidden-tile n0; the 4 gate B-tiles
// (4KB per K-step) are staged into double-buffered LDS with gfx1250 async
// LDS loads (global_load_async_to_lds_b128, ASYNCcnt) so B moves L2->LDS
// with no VGPR transit and overlaps the WMMA burst. All four B tiles are
// materialized in registers before the WMMA burst so the 4 WMMAs issue
// back-to-back behind a single s_wait_dscnt.
// ---------------------------------------------------------------------------

#define SEQLEN 128
#define BATCH  256
#define HDIM   768
#define WELEMS (4 * HDIM * HDIM)                 // 2359296 elements per w_hh
#define LROW   40                                // LDS row pitch in bf16 (80B)

typedef __attribute__((ext_vector_type(16))) __bf16 v16bf;
typedef __attribute__((ext_vector_type(8)))  float  v8f;

union BfVec { v16bf v; uint32_t u[8]; };
union F8    { v8f  v; float   f[8]; };

__device__ __forceinline__ uint32_t pk_bf16_rne(float a, float b) {
  uint32_t ua = __float_as_uint(a);
  uint32_t ub = __float_as_uint(b);
  ua += 0x7FFFu + ((ua >> 16) & 1u);   // round-to-nearest-even
  ub += 0x7FFFu + ((ub >> 16) & 1u);
  return (ua >> 16) | (ub & 0xFFFF0000u);
}

__device__ __forceinline__ uint16_t bf16_rne(float a) {
  uint32_t ua = __float_as_uint(a);
  ua += 0x7FFFu + ((ua >> 16) & 1u);
  return (uint16_t)(ua >> 16);
}

__device__ __forceinline__ float fsig(float x) {
  return 1.0f / (1.0f + __expf(-x));
}

__device__ __forceinline__ float ftanh(float x) {
  x = fminf(fmaxf(x, -15.0f), 15.0f);
  float e = __expf(-2.0f * x);
  return (1.0f - e) / (1.0f + e);
}

__device__ __forceinline__ void wait_async0() {
#if __has_builtin(__builtin_amdgcn_s_wait_asynccnt)
  __builtin_amdgcn_s_wait_asynccnt(0);
#else
  asm volatile("s_wait_asynccnt 0x0" ::: "memory");
#endif
}

// Async copy of 16B from global to LDS (gfx1250 GLOBAL_LOAD_ASYNC_TO_LDS_B128,
// tracked by ASYNCcnt). lds_addr = low 32 bits of the flat shared pointer
// (ISA 10.2: LDS aperture addresses truncate to addr[31:0]).
__device__ __forceinline__ void async_b128_to_lds(uint32_t lds_addr,
                                                  const void* gsrc) {
  asm volatile("global_load_async_to_lds_b128 %0, %1, off"
               :: "v"(lds_addr), "v"(gsrc) : "memory");
}

// Fused LSTM step. Grid: 96 blocks x 256 threads (8 waves).
// blockIdx -> (n-tile, m-half): all 8 waves share hidden columns n0..n0+15
// and cover batch rows mhalf*128 .. +127 (16 rows per wave).
// Each wave accumulates the 16x16 tile of all four gates, then does the
// elementwise gate math and updates c (f32, in place) and h (bf16).
template <bool WBF16>
__global__ __launch_bounds__(256)
void lstm_step(const uint16_t* __restrict__ h_in,    // [B,H] bf16 (prev h)
               uint16_t*       __restrict__ h_out,   // [B,H] bf16 (new h)
               float*          __restrict__ c_st,    // [B,H] f32  (in-place)
               const uint16_t* __restrict__ w_bf,    // [4H,H] bf16 (if WBF16)
               const float*    __restrict__ w_f32,   // [4H,H] f32  (if !WBF16)
               const float*    __restrict__ b_ih,    // [4H]
               const float*    __restrict__ b_hh,    // [4H]
               const float*    __restrict__ w_ih,    // [4H], or null
               const float*    __restrict__ x_t,     // [B],  or null
               int use_x)
{
  const int nb    = blockIdx.x >> 1;          // 0..47 hidden tile
  const int mhalf = blockIdx.x & 1;           // 0..1  batch half
  const int n0    = nb << 4;
  const int wave  = threadIdx.x >> 5;         // 0..7
  const int m0    = (mhalf << 7) + (wave << 4);
  const int lane  = threadIdx.x & 31;
  const int g     = lane >> 4;                // lane half-group
  const int l     = lane & 15;

  // Double-buffered B stage: [buf][(gate*16 + col)*LROW + k], 80B row pitch
  // so the 64-row strided tile reads spread across the 64 LDS banks.
  __shared__ uint16_t bsh[2][64 * LROW];      // 2 x 5120 B

  // Copy stage: thread t moves one 16B (8-element) chunk per K-step.
  const int t     = threadIdx.x;
  const int crow  = t >> 2;                   // 0..63 = gate*16 + col
  const int cchk  = t & 3;                    // 0..3
  const int cgate = crow >> 4;
  const int ccol  = crow & 15;
  const size_t csrc = (size_t)(cgate * HDIM + n0 + ccol) * HDIM + cchk * 8;
  const int    cdst = crow * LROW + cchk * 8;
  const uint32_t lds_dst0 = (uint32_t)(uintptr_t)&bsh[0][cdst];
  const uint32_t lds_dst1 = (uint32_t)(uintptr_t)&bsh[1][cdst];

  // A-matrix 16x32 bf16 (ISA 7.12.2): lanes 0-15 hold row M=l with
  // K = {k0..k0+7, k0+16..k0+23}; lanes 16-31 the complementary K bytes.
  const uint16_t* arow = h_in + (size_t)(m0 + l) * HDIM + 8 * g;

  F8 acc[4];
  #pragma unroll
  for (int gg = 0; gg < 4; ++gg) { v8f z = {}; acc[gg].v = z; }

  // Prologue: stage K-block 0 into buffer 0.
  if (WBF16) {
    async_b128_to_lds(lds_dst0, w_bf + csrc);
    wait_async0();
  } else {
    const float4* s = reinterpret_cast<const float4*>(w_f32 + csrc);
    float4 s0 = s[0], s1 = s[1];
    uint4 v;
    v.x = pk_bf16_rne(s0.x, s0.y); v.y = pk_bf16_rne(s0.z, s0.w);
    v.z = pk_bf16_rne(s1.x, s1.y); v.w = pk_bf16_rne(s1.z, s1.w);
    *reinterpret_cast<uint4*>(&bsh[0][cdst]) = v;
  }
  __syncthreads();

  int buf = 0;
  for (int k0 = 0; k0 < HDIM; k0 += 32, buf ^= 1) {
    const bool have_next = (k0 + 32) < HDIM;   // uniform

    // Issue next K-block's stage early so it overlaps the WMMA burst.
    float4 nF0 = {0, 0, 0, 0}, nF1 = {0, 0, 0, 0};
    if (have_next) {
      if (WBF16) {
        async_b128_to_lds(buf ? lds_dst0 : lds_dst1, w_bf + csrc + k0 + 32);
      } else {
        const float4* s = reinterpret_cast<const float4*>(w_f32 + csrc + k0 + 32);
        nF0 = s[0]; nF1 = s[1];
      }
    }

    // A tile from global (each wave owns distinct batch rows).
    BfVec A;
    const uint4* pa = reinterpret_cast<const uint4*>(arow + k0);
    uint4 a0 = pa[0];           // K = k0 + 8g + 0..7
    uint4 a1 = pa[2];           // K = k0 + 8g + 16..23
    A.u[0] = a0.x; A.u[1] = a0.y; A.u[2] = a0.z; A.u[3] = a0.w;
    A.u[4] = a1.x; A.u[5] = a1.y; A.u[6] = a1.z; A.u[7] = a1.w;

    // Materialize all four gate B tiles first (8 ds_load_b128 in one clause,
    // one dscnt wait), then issue the 4 WMMAs back-to-back.
    BfVec Bt[4];
    #pragma unroll
    for (int gg = 0; gg < 4; ++gg) {
      const uint4* pb = reinterpret_cast<const uint4*>(
          &bsh[buf][(gg * 16 + l) * LROW + g * 8]);
      uint4 b0 = pb[0], b1 = pb[2];
      Bt[gg].u[0] = b0.x; Bt[gg].u[1] = b0.y;
      Bt[gg].u[2] = b0.z; Bt[gg].u[3] = b0.w;
      Bt[gg].u[4] = b1.x; Bt[gg].u[5] = b1.y;
      Bt[gg].u[6] = b1.z; Bt[gg].u[7] = b1.w;
    }
    #pragma unroll
    for (int gg = 0; gg < 4; ++gg) {
      acc[gg].v = __builtin_amdgcn_wmma_f32_16x16x32_bf16(
          /*neg_a=*/false, A.v, /*neg_b=*/false, Bt[gg].v,
          /*c_mod=*/(short)0, acc[gg].v,
          /*reuse_a=*/false, /*reuse_b=*/false);
    }

    // Make next buffer's contents resident before the publishing barrier.
    if (have_next) {
      if (WBF16) {
        wait_async0();
      } else {
        uint4 v;
        v.x = pk_bf16_rne(nF0.x, nF0.y); v.y = pk_bf16_rne(nF0.z, nF0.w);
        v.z = pk_bf16_rne(nF1.x, nF1.y); v.w = pk_bf16_rne(nF1.z, nF1.w);
        *reinterpret_cast<uint4*>(&bsh[buf ^ 1][cdst]) = v;
      }
    }
    __syncthreads();   // publishes next buffer / protects current from reuse
  }

  // Epilogue. C/D layout: VGPR r, lanes 0-15 -> (M=r, N=l); lanes 16-31 ->
  // (M=r+8, N=l). This lane owns column n = n0+l, rows b = m0 + 8g + r.
  const int n = n0 + l;
  float bias[4], wih[4];
  #pragma unroll
  for (int gg = 0; gg < 4; ++gg) {
    bias[gg] = b_ih[gg * HDIM + n] + b_hh[gg * HDIM + n];
    wih[gg]  = use_x ? w_ih[gg * HDIM + n] : 0.0f;   // uniform branch
  }
  float xv[8];
  #pragma unroll
  for (int r = 0; r < 8; ++r)
    xv[r] = use_x ? x_t[m0 + 8 * g + r] : 0.0f;

  #pragma unroll
  for (int r = 0; r < 8; ++r) {
    float pi = acc[0].f[r] + bias[0] + xv[r] * wih[0];
    float pf = acc[1].f[r] + bias[1] + xv[r] * wih[1];
    float pg = acc[2].f[r] + bias[2] + xv[r] * wih[2];
    float po = acc[3].f[r] + bias[3] + xv[r] * wih[3];
    float ig = fsig(pi), fg = fsig(pf), gt = ftanh(pg), og = fsig(po);
    size_t idx = (size_t)(m0 + 8 * g + r) * HDIM + n;
    float cn = fg * c_st[idx] + ig * gt;
    c_st[idx]  = cn;
    h_out[idx] = bf16_rne(og * ftanh(cn));
  }
}

// Convert a weight matrix f32 -> bf16 (RNE), 4 elements per thread.
__global__ void cvt_f32_to_bf16(const float* __restrict__ src,
                                uint16_t* __restrict__ dst, int n4) {
  int i = blockIdx.x * blockDim.x + threadIdx.x;
  if (i < n4) {
    float4 s = reinterpret_cast<const float4*>(src)[i];
    reinterpret_cast<uint2*>(dst)[i] =
        make_uint2(pk_bf16_rne(s.x, s.y), pk_bf16_rne(s.z, s.w));
  }
}

__global__ void init_state(uint32_t* __restrict__ ws, int n_words) {
  int i = blockIdx.x * blockDim.x + threadIdx.x;
  if (i < n_words) ws[i] = 0u;
}

// softmax over a size-1 axis is identically 1.0
__global__ void write_ones(float* __restrict__ out, int n) {
  int i = blockIdx.x * blockDim.x + threadIdx.x;
  if (i < n) out[i] = 1.0f;
}

extern "C" void kernel_launch(void* const* d_in, const int* in_sizes, int n_in,
                              void* d_out, int out_size, void* d_ws, size_t ws_size,
                              hipStream_t stream) {
  (void)in_sizes; (void)n_in;
  const float* x         = (const float*)d_in[0];   // [128,256,1]
  const float* w_ih_enc  = (const float*)d_in[1];   // [4H,1]
  const float* w_hh_enc  = (const float*)d_in[2];   // [4H,H]
  const float* b_ih_enc  = (const float*)d_in[3];
  const float* b_hh_enc  = (const float*)d_in[4];
  // d_in[5] w_ih_dec: decoder input is all-zeros -> never contributes
  const float* w_hh_dec  = (const float*)d_in[6];
  const float* b_ih_dec  = (const float*)d_in[7];
  const float* b_hh_dec  = (const float*)d_in[8];
  // d_in[9] w_lin / d_in[10] b_lin: dead — softmax over a size-1 dim is 1.0

  const int HN = BATCH * HDIM;                      // 196608
  const size_t state_bytes = (size_t)HN * 2 * 2 + (size_t)HN * 4;  // 1.5 MB
  const size_t wbf_bytes   = (size_t)WELEMS * 2;                   // 4.5 MB each
  const bool   fast        = ws_size >= state_bytes + 2 * wbf_bytes;

  // Workspace layout: [hbuf0 bf16][hbuf1 bf16][c f32][wenc bf16][wdec bf16]
  uint16_t* hbuf0 = (uint16_t*)d_ws;
  uint16_t* hbuf1 = hbuf0 + HN;
  float*    c_st  = (float*)(hbuf1 + HN);
  uint16_t* wenc  = (uint16_t*)(c_st + HN);
  uint16_t* wdec  = wenc + WELEMS;

  init_state<<<(int)((state_bytes / 4 + 255) / 256), 256, 0, stream>>>(
      (uint32_t*)d_ws, (int)(state_bytes / 4));

  if (fast) {
    const int n4 = WELEMS / 4;
    cvt_f32_to_bf16<<<(n4 + 255) / 256, 256, 0, stream>>>(w_hh_enc, wenc, n4);
    cvt_f32_to_bf16<<<(n4 + 255) / 256, 256, 0, stream>>>(w_hh_dec, wdec, n4);
  }

  uint16_t* hin = hbuf0;
  uint16_t* hout = hbuf1;
  // Encoder: 128 sequential fused steps (graph replay amortizes launches).
  for (int t = 0; t < SEQLEN; ++t) {
    if (fast)
      lstm_step<true><<<96, 256, 0, stream>>>(hin, hout, c_st, wenc, nullptr,
                                              b_ih_enc, b_hh_enc, w_ih_enc,
                                              x + (size_t)t * BATCH, 1);
    else
      lstm_step<false><<<96, 256, 0, stream>>>(hin, hout, c_st, nullptr, w_hh_enc,
                                               b_ih_enc, b_hh_enc, w_ih_enc,
                                               x + (size_t)t * BATCH, 1);
    uint16_t* tmp = hin; hin = hout; hout = tmp;
  }
  // Decoder: input path is identically zero -> bias-only contribution.
  for (int t = 0; t < SEQLEN; ++t) {
    if (fast)
      lstm_step<true><<<96, 256, 0, stream>>>(hin, hout, c_st, wdec, nullptr,
                                              b_ih_dec, b_hh_dec, nullptr,
                                              nullptr, 0);
    else
      lstm_step<false><<<96, 256, 0, stream>>>(hin, hout, c_st, nullptr, w_hh_dec,
                                               b_ih_dec, b_hh_dec, nullptr,
                                               nullptr, 0);
    uint16_t* tmp = hin; hin = hout; hout = tmp;
  }

  write_ones<<<(out_size + 255) / 256, 256, 0, stream>>>((float*)d_out, out_size);
}
